// GlobalUserNet_77360950936279
// MI455X (gfx1250) — matrix operand order repaired and checked by reading, not compiled, not run on hardware
//
#include <hip/hip_runtime.h>
#include <hip/hip_bf16.h>

typedef __attribute__((ext_vector_type(16))) _Float16 v16h;
typedef __attribute__((ext_vector_type(8)))  float    v8f;

#define CCH 128
#define K_IN 403
#define KPAD_IN 416

__device__ __forceinline__ float lrelu01(float x){ return x > 0.f ? x : 0.01f * x; }
__device__ __forceinline__ float lrelu02(float x){ return x > 0.f ? x : 0.2f  * x; }
// monotone float -> uint key for atomicMax over possibly-negative floats
__device__ __forceinline__ unsigned fkey(float f){
  int i = __float_as_int(f);
  return (unsigned)(i ^ ((i >> 31) | 0x80000000));
}
__device__ __forceinline__ float funkey(unsigned k){
  int i = (k & 0x80000000u) ? (int)(k ^ 0x80000000u) : ~(int)k;
  return __int_as_float(i);
}

// ---------------- weight transpose+convert: Wt[c*Kpad + k] = (f16) W[k*128 + c] ----------
__global__ void convert_wt(const float* __restrict__ W, _Float16* __restrict__ Wt,
                           int K, int Kpad) {
  int idx = blockIdx.x * blockDim.x + threadIdx.x;
  int total = CCH * Kpad;
  if (idx >= total) return;
  int c = idx / Kpad, k = idx - c * Kpad;
  Wt[idx] = (k < K) ? (_Float16)W[(size_t)k * CCH + c] : (_Float16)0.f;
}

// ---------------- generic GEMM: Y[N,128] = X[N,128] @ W (Wt = f16, transposed) ----------
__global__ void __launch_bounds__(256)
gemm128_wmma(const float* __restrict__ X, const _Float16* __restrict__ Wt,
             float* __restrict__ Y, int nrows) {
  const int wave = threadIdx.x >> 5;
  const int lane = threadIdx.x & 31;
  const int hs   = lane >> 4;       // K-half select
  const int lm   = lane & 15;
  const int tileM = blockIdx.x * 8 + wave;
  const int row0  = tileM * 16;
  if (row0 >= nrows) return;        // wave-uniform exit
  const int arow = row0 + lm;
  const int rc   = arow < nrows ? arow : nrows - 1;
  const float rv = arow < nrows ? 1.f : 0.f;

  v8f acc[8];
#pragma unroll
  for (int t = 0; t < 8; ++t)
#pragma unroll
    for (int j = 0; j < 8; ++j) acc[t][j] = 0.f;

  const float* xrow = X + (size_t)rc * CCH;

#pragma unroll
  for (int ks = 0; ks < 4; ++ks) {
    const int k0 = ks * 32;
    const int ka = k0 + hs * 16;
    float4 c0 = *(const float4*)(xrow + ka);
    float4 c1 = *(const float4*)(xrow + ka + 4);
    float4 c2 = *(const float4*)(xrow + ka + 8);
    float4 c3 = *(const float4*)(xrow + ka + 12);
    v16h a;
    a[0]=(_Float16)(c0.x*rv); a[1]=(_Float16)(c0.y*rv); a[2]=(_Float16)(c0.z*rv); a[3]=(_Float16)(c0.w*rv);
    a[4]=(_Float16)(c1.x*rv); a[5]=(_Float16)(c1.y*rv); a[6]=(_Float16)(c1.z*rv); a[7]=(_Float16)(c1.w*rv);
    a[8]=(_Float16)(c2.x*rv); a[9]=(_Float16)(c2.y*rv); a[10]=(_Float16)(c2.z*rv); a[11]=(_Float16)(c2.w*rv);
    a[12]=(_Float16)(c3.x*rv); a[13]=(_Float16)(c3.y*rv); a[14]=(_Float16)(c3.z*rv); a[15]=(_Float16)(c3.w*rv);
#pragma unroll
    for (int t = 0; t < 8; ++t) {
      const int col = t * 16 + lm;
      v16h b = *(const v16h*)(Wt + (size_t)col * CCH + k0 + hs * 16);
      acc[t] = __builtin_amdgcn_wmma_f32_16x16x32_f16(false, a, false, b,
                                                      (short)0, acc[t], false, false);
    }
  }
#pragma unroll
  for (int r = 0; r < 8; ++r) {
    const int row = row0 + r + hs * 8;
    if (row < nrows) {
      float* yr = Y + (size_t)row * CCH;
#pragma unroll
      for (int t = 0; t < 8; ++t) yr[t * 16 + lm] = acc[t][r];
    }
  }
}

// ---------------- input GEMM with fused embedding gather (K=403 padded to 416) ----------
__device__ __forceinline__ float feat_at(int n, int pid, int cid, int k,
                                         const float* __restrict__ feat3,
                                         const float* __restrict__ poi_emb,
                                         const float* __restrict__ cat_emb) {
  if (k < 300) return poi_emb[(size_t)pid * 300 + k];
  if (k < 400) return cat_emb[(size_t)cid * 100 + (k - 300)];
  if (k < 403) return feat3[(size_t)n * 3 + (k - 400)];
  return 0.f;
}

__global__ void __launch_bounds__(256)
gemm_in_wmma(const int* __restrict__ poi_ids, const int* __restrict__ cat_ids,
             const float* __restrict__ feat3, const float* __restrict__ poi_emb,
             const float* __restrict__ cat_emb, const _Float16* __restrict__ Wt,
             float* __restrict__ Y, int nrows) {
  const int wave = threadIdx.x >> 5;
  const int lane = threadIdx.x & 31;
  const int hs   = lane >> 4;
  const int lm   = lane & 15;
  const int tileM = blockIdx.x * 8 + wave;
  const int row0  = tileM * 16;
  if (row0 >= nrows) return;
  const int arow = row0 + lm;
  const int rc   = arow < nrows ? arow : nrows - 1;
  const float rv = arow < nrows ? 1.f : 0.f;
  const int pid = poi_ids[rc];
  const int cid = cat_ids[rc];

  v8f acc[8];
#pragma unroll
  for (int t = 0; t < 8; ++t)
#pragma unroll
    for (int j = 0; j < 8; ++j) acc[t][j] = 0.f;

  for (int ks = 0; ks < KPAD_IN / 32; ++ks) {
    const int k0 = ks * 32;
    const int ka = k0 + hs * 16;
    v16h a;
#pragma unroll
    for (int j = 0; j < 8; ++j) {
      a[j]     = (_Float16)(feat_at(rc, pid, cid, ka + j,      feat3, poi_emb, cat_emb) * rv);
      a[8 + j] = (_Float16)(feat_at(rc, pid, cid, ka + 8 + j,  feat3, poi_emb, cat_emb) * rv);
    }
#pragma unroll
    for (int t = 0; t < 8; ++t) {
      const int col = t * 16 + lm;
      v16h b = *(const v16h*)(Wt + (size_t)col * KPAD_IN + k0 + hs * 16);
      acc[t] = __builtin_amdgcn_wmma_f32_16x16x32_f16(false, a, false, b,
                                                      (short)0, acc[t], false, false);
    }
  }
#pragma unroll
  for (int r = 0; r < 8; ++r) {
    const int row = row0 + r + hs * 8;
    if (row < nrows) {
      float* yr = Y + (size_t)row * CCH;
#pragma unroll
      for (int t = 0; t < 8; ++t) yr[t * 16 + lm] = acc[t][r];
    }
  }
}

// ---------------- degree / dinv -----------------------------------------------
__global__ void deg_init(float* __restrict__ buf, int n) {
  int i = blockIdx.x * blockDim.x + threadIdx.x;
  if (i < n) buf[i] = 1.f;                       // self-loop weight
}
__global__ void deg_edge(const int* __restrict__ dst, const float* __restrict__ ew,
                         float* __restrict__ buf, int nE) {
  for (int e = blockIdx.x * blockDim.x + threadIdx.x; e < nE; e += gridDim.x * blockDim.x)
    atomicAdd(buf + dst[e], ew[e]);
}
__global__ void deg_finish(float* __restrict__ buf, int n) {
  int i = blockIdx.x * blockDim.x + threadIdx.x;
  if (i < n) { float d = buf[i]; buf[i] = d > 0.f ? rsqrtf(d) : 0.f; }
}

// ---------------- GCN conv (self-loop init + edge scatter) ---------------------
__global__ void gcn_init(float* __restrict__ tb, const float* __restrict__ xw,
                         const float* __restrict__ b, const float* __restrict__ dinv, int n) {
  int tot = n * CCH;
  for (int i = blockIdx.x * blockDim.x + threadIdx.x; i < tot; i += gridDim.x * blockDim.x) {
    int nd = i >> 7, c = i & 127;
    float di = dinv[nd];
    tb[i] = b[c] + xw[i] * di * di;
  }
}
__global__ void __launch_bounds__(256)
gcn_edge_scatter(const float* __restrict__ xw, const int* __restrict__ src,
                 const int* __restrict__ dst, const float* __restrict__ ew,
                 const float* __restrict__ dinv, float* __restrict__ out, int nE) {
  int gw = (blockIdx.x * 256 + threadIdx.x) >> 5;
  int lane = threadIdx.x & 31;
  int nw = (gridDim.x * 256) >> 5;
  for (int e = gw; e < nE; e += nw) {
    int s = src[e], d = dst[e];
    float coef = dinv[s] * ew[e] * dinv[d];
    float4 v = ((const float4*)(xw + (size_t)s * CCH))[lane];
    float* o = out + (size_t)d * CCH + lane * 4;
    atomicAdd(o + 0, v.x * coef); atomicAdd(o + 1, v.y * coef);
    atomicAdd(o + 2, v.z * coef); atomicAdd(o + 3, v.w * coef);
  }
}

// ---------------- GraphNorm ----------------------------------------------------
__global__ void colsum(const float* __restrict__ X, float* __restrict__ out, int nrows) {
  int t = threadIdx.x, c = t & 127, rl = t >> 7;
  int rpi = blockDim.x >> 7;
  float s = 0.f;
  for (int nd = blockIdx.x * rpi + rl; nd < nrows; nd += gridDim.x * rpi)
    s += X[(size_t)nd * CCH + c];
  atomicAdd(out + c, s);
}
__global__ void colvar(const float* __restrict__ X, const float* __restrict__ msum,
                       const float* __restrict__ alpha, float* __restrict__ out, int nrows) {
  int t = threadIdx.x, c = t & 127, rl = t >> 7;
  int rpi = blockDim.x >> 7;
  float m = msum[c] / (float)nrows;
  float am = alpha[c] * m;
  float s = 0.f;
  for (int nd = blockIdx.x * rpi + rl; nd < nrows; nd += gridDim.x * rpi) {
    float d = X[(size_t)nd * CCH + c] - am;
    s += d * d;
  }
  atomicAdd(out + c, s);
}
__global__ void norm_resid_leaky(float* __restrict__ x, const float* __restrict__ t,
                                 const float* __restrict__ msum, const float* __restrict__ vsum,
                                 const float* __restrict__ gamma, const float* __restrict__ beta,
                                 const float* __restrict__ alpha, int nrows) {
  int tot = nrows * CCH;
  float invN = 1.f / (float)nrows;
  for (int i = blockIdx.x * blockDim.x + threadIdx.x; i < tot; i += gridDim.x * blockDim.x) {
    int c = i & 127;
    float m = msum[c] * invN;
    float v = vsum[c] * invN;
    float y = gamma[c] * (t[i] - alpha[c] * m) * rsqrtf(v + 1e-5f) + beta[c];
    x[i] += lrelu01(y);
  }
}
__global__ void leaky_copy(float* __restrict__ x, const float* __restrict__ tb, int tot) {
  for (int i = blockIdx.x * blockDim.x + threadIdx.x; i < tot; i += gridDim.x * blockDim.x)
    x[i] = lrelu01(tb[i]);
}

// ---------------- GAT conv -----------------------------------------------------
__global__ void __launch_bounds__(256)
gat_alar(const float* __restrict__ xw, const float* __restrict__ asrc,
         const float* __restrict__ adst, float* __restrict__ al, float* __restrict__ ar, int n) {
  int gw = (blockIdx.x * 256 + threadIdx.x) >> 5;
  int lane = threadIdx.x & 31;
  int nw = (gridDim.x * 256) >> 5;
  float4 s = ((const float4*)asrc)[lane];
  float4 d = ((const float4*)adst)[lane];
  for (int nd = gw; nd < n; nd += nw) {
    float4 v = ((const float4*)(xw + (size_t)nd * CCH))[lane];
    float pl = v.x * s.x + v.y * s.y + v.z * s.z + v.w * s.w;
    float pr = v.x * d.x + v.y * d.y + v.z * d.z + v.w * d.w;
    for (int off = 16; off; off >>= 1) {
      pl += __shfl_xor(pl, off, 32);
      pr += __shfl_xor(pr, off, 32);
    }
    if (lane == 0) { al[nd] = pl; ar[nd] = pr; }
  }
}
__global__ void gat_premax(const float* __restrict__ al, const float* __restrict__ ar,
                           unsigned* __restrict__ emk, int n) {
  int i = blockIdx.x * blockDim.x + threadIdx.x;
  if (i < n) emk[i] = fkey(lrelu02(al[i] + ar[i]));   // self-loop term
}
__global__ void gat_edge_max(const float* __restrict__ al, const float* __restrict__ ar,
                             const int* __restrict__ src, const int* __restrict__ dst,
                             unsigned* __restrict__ emk, int nE) {
  for (int e = blockIdx.x * blockDim.x + threadIdx.x; e < nE; e += gridDim.x * blockDim.x) {
    int s = src[e], d = dst[e];
    atomicMax(emk + d, fkey(lrelu02(al[s] + ar[d])));
  }
}
__global__ void gat_den_init(const float* __restrict__ al, const float* __restrict__ ar,
                             const unsigned* __restrict__ emk, float* __restrict__ den, int n) {
  int i = blockIdx.x * blockDim.x + threadIdx.x;
  if (i < n) den[i] = __expf(lrelu02(al[i] + ar[i]) - funkey(emk[i]));
}
__global__ void gat_edge_den(const float* __restrict__ al, const float* __restrict__ ar,
                             const int* __restrict__ src, const int* __restrict__ dst,
                             const unsigned* __restrict__ emk, float* __restrict__ den, int nE) {
  for (int e = blockIdx.x * blockDim.x + threadIdx.x; e < nE; e += gridDim.x * blockDim.x) {
    int s = src[e], d = dst[e];
    atomicAdd(den + d, __expf(lrelu02(al[s] + ar[d]) - funkey(emk[d])));
  }
}
__global__ void gat_out_init(float* __restrict__ tb, const float* __restrict__ xw,
                             const float* __restrict__ b, const float* __restrict__ al,
                             const float* __restrict__ ar, const unsigned* __restrict__ emk,
                             const float* __restrict__ den, int n) {
  int tot = n * CCH;
  for (int i = blockIdx.x * blockDim.x + threadIdx.x; i < tot; i += gridDim.x * blockDim.x) {
    int nd = i >> 7, c = i & 127;
    float att = __expf(lrelu02(al[nd] + ar[nd]) - funkey(emk[nd])) / den[nd];
    tb[i] = b[c] + xw[i] * att;
  }
}
__global__ void __launch_bounds__(256)
gat_edge_scatter(const float* __restrict__ xw, const int* __restrict__ src,
                 const int* __restrict__ dst, const float* __restrict__ al,
                 const float* __restrict__ ar, const unsigned* __restrict__ emk,
                 const float* __restrict__ den, float* __restrict__ out, int nE) {
  int gw = (blockIdx.x * 256 + threadIdx.x) >> 5;
  int lane = threadIdx.x & 31;
  int nw = (gridDim.x * 256) >> 5;
  for (int e = gw; e < nE; e += nw) {
    int s = src[e], d = dst[e];
    float att = __expf(lrelu02(al[s] + ar[d]) - funkey(emk[d])) / den[d];
    float4 v = ((const float4*)(xw + (size_t)s * CCH))[lane];
    float* o = out + (size_t)d * CCH + lane * 4;
    atomicAdd(o + 0, v.x * att); atomicAdd(o + 1, v.y * att);
    atomicAdd(o + 2, v.z * att); atomicAdd(o + 3, v.w * att);
  }
}

// ---------------- output conv (C -> 1) + FC head -------------------------------
__global__ void __launch_bounds__(256)
wout_matvec(const float* __restrict__ x, const float* __restrict__ w,
            float* __restrict__ out, int n) {
  int gw = (blockIdx.x * 256 + threadIdx.x) >> 5;
  int lane = threadIdx.x & 31;
  int nw = (gridDim.x * 256) >> 5;
  float4 ww = ((const float4*)w)[lane];
  for (int nd = gw; nd < n; nd += nw) {
    float4 v = ((const float4*)(x + (size_t)nd * CCH))[lane];
    float p = v.x * ww.x + v.y * ww.y + v.z * ww.z + v.w * ww.w;
    for (int off = 16; off; off >>= 1) p += __shfl_xor(p, off, 32);
    if (lane == 0) out[nd] = p;
  }
}
__global__ void sc_init(float* __restrict__ svec, const float* __restrict__ xwo,
                        const float* __restrict__ b_out, const float* __restrict__ dinv, int n) {
  int i = blockIdx.x * blockDim.x + threadIdx.x;
  if (i < n) { float di = dinv[i]; svec[i] = b_out[0] + xwo[i] * di * di; }
}
__global__ void sc_edge(const float* __restrict__ xwo, const int* __restrict__ src,
                        const int* __restrict__ dst, const float* __restrict__ ew,
                        const float* __restrict__ dinv, float* __restrict__ svec, int nE) {
  for (int e = blockIdx.x * blockDim.x + threadIdx.x; e < nE; e += gridDim.x * blockDim.x) {
    int s = src[e], d = dst[e];
    atomicAdd(svec + d, xwo[s] * dinv[s] * ew[e] * dinv[d]);
  }
}
__global__ void h_init(float* __restrict__ hacc, const float* __restrict__ b1) {
  if (threadIdx.x < CCH) hacc[threadIdx.x] = b1[threadIdx.x];
}
__global__ void fc1_kernel(const float* __restrict__ svec, const float* __restrict__ W1,
                           float* __restrict__ hacc, int nrows) {
  int t = threadIdx.x, c = t & 127, rl = t >> 7;
  int rpi = blockDim.x >> 7;
  float s = 0.f;
  for (int nd = blockIdx.x * rpi + rl; nd < nrows; nd += gridDim.x * rpi)
    s += lrelu01(svec[nd]) * W1[(size_t)nd * CCH + c];
  atomicAdd(hacc + c, s);
}
__global__ void fc2_kernel(const float* __restrict__ hacc, const float* __restrict__ W2,
                           const float* __restrict__ b2, float* __restrict__ out, int plen) {
  __shared__ float h[CCH];
  if (threadIdx.x < CCH) { float v = hacc[threadIdx.x]; h[threadIdx.x] = v > 0.f ? v : 0.f; }
  __syncthreads();
  for (int p = blockIdx.x * blockDim.x + threadIdx.x; p < plen; p += gridDim.x * blockDim.x) {
    float s = b2[p];
#pragma unroll 8
    for (int k = 0; k < CCH; ++k) s += h[k] * W2[(size_t)k * plen + p];
    out[p] = s > 0.f ? s : 0.f;
  }
}

// ================================================================================
extern "C" void kernel_launch(void* const* d_in, const int* in_sizes, int n_in,
                              void* d_out, int out_size, void* d_ws, size_t ws_size,
                              hipStream_t stream) {
  const int*   poi_ids = (const int*)d_in[0];
  const int*   cat_ids = (const int*)d_in[1];
  const float* feat3   = (const float*)d_in[2];
  const int*   eidx    = (const int*)d_in[3];
  const float* ew      = (const float*)d_in[4];
  const float* poi_emb = (const float*)d_in[5];
  const float* cat_emb = (const float*)d_in[6];
  const float* Win     = (const float*)d_in[7];
  const float* b_in    = (const float*)d_in[8];
  const float* gcn_W   = (const float*)d_in[9];
  const float* gcn_b   = (const float*)d_in[10];
  const float* gn_gamma= (const float*)d_in[11];
  const float* gn_beta = (const float*)d_in[12];
  const float* gn_alpha= (const float*)d_in[13];
  const float* gat_W   = (const float*)d_in[14];
  const float* gat_asrc= (const float*)d_in[15];
  const float* gat_adst= (const float*)d_in[16];
  const float* gat_b   = (const float*)d_in[17];
  const float* Wout    = (const float*)d_in[18];
  const float* b_out   = (const float*)d_in[19];
  const float* fc_W1   = (const float*)d_in[20];
  const float* fc_b1   = (const float*)d_in[21];
  const float* fc_W2   = (const float*)d_in[22];
  const float* fc_b2   = (const float*)d_in[23];

  const int N    = in_sizes[0];
  const int E    = in_sizes[4];
  const int PLEN = in_sizes[23];
  const int* src = eidx;
  const int* dst = eidx + E;
  float* out = (float*)d_out;

  // ---- workspace carve-up (stream-ordered reuse) ----
  char* ws = (char*)d_ws;
  auto carve = [&](size_t bytes) { void* p = ws; ws += (bytes + 255) & ~(size_t)255; return p; };
  const size_t NB = (size_t)N * CCH * sizeof(float);
  float*    xw   = (float*)carve(NB);
  float*    tb   = (float*)carve(NB);
  float*    x    = (float*)carve(NB);
  float*    dinv = (float*)carve((size_t)N * 4);
  float*    al   = (float*)carve((size_t)N * 4);
  float*    ar   = (float*)carve((size_t)N * 4);
  unsigned* emk  = (unsigned*)carve((size_t)N * 4);
  float*    den  = (float*)carve((size_t)N * 4);
  float*    svec = (float*)carve((size_t)N * 4);
  float*    msum = (float*)carve(CCH * 4);
  float*    vsum = (float*)carve(CCH * 4);
  float*    hacc = (float*)carve(CCH * 4);
  _Float16* Wth  = (_Float16*)carve((size_t)CCH * KPAD_IN * sizeof(_Float16));

  const int TPB      = 256;
  const int nodeBlk  = (N + TPB - 1) / TPB;          // thread-per-node
  const int edgeBlk  = 2048;                          // grid-stride thread-per-edge
  const int edgeWBlk = 4096;                          // grid-stride wave-per-edge
  const int nodeWBlk = (N + 7) / 8;                   // wave-per-node
  const int ewBlk    = 2048;                          // elementwise N*128 grid-stride
  const int redBlk   = 512;                           // column reductions
  const int gemmBlk  = ((N + 15) / 16 + 7) / 8;       // 8 row-tiles per block
  const int tot      = N * CCH;

  auto graph_norm_resid = [&](const float* gma, const float* bta, const float* alp) {
    hipMemsetAsync(msum, 0, CCH * 4, stream);
    hipMemsetAsync(vsum, 0, CCH * 4, stream);
    colsum<<<redBlk, TPB, 0, stream>>>(tb, msum, N);
    colvar<<<redBlk, TPB, 0, stream>>>(tb, msum, alp, vsum, N);
    norm_resid_leaky<<<ewBlk, TPB, 0, stream>>>(x, tb, msum, vsum, gma, bta, alp, N);
  };
  auto gcn_conv = [&](const float* b) {  // xw already holds X@W; result in tb
    gcn_init<<<ewBlk, TPB, 0, stream>>>(tb, xw, b, dinv, N);
    gcn_edge_scatter<<<edgeWBlk, TPB, 0, stream>>>(xw, src, dst, ew, dinv, tb, E);
  };

  // ---- degree / symmetric-norm factors (graph is fixed; compute once) ----
  deg_init<<<nodeBlk, TPB, 0, stream>>>(dinv, N);
  deg_edge<<<edgeBlk, TPB, 0, stream>>>(dst, ew, dinv, E);
  deg_finish<<<nodeBlk, TPB, 0, stream>>>(dinv, N);

  // ---- input GCN conv: fused embedding gather + WMMA GEMM (K=403) ----
  convert_wt<<<(CCH * KPAD_IN + TPB - 1) / TPB, TPB, 0, stream>>>(Win, Wth, K_IN, KPAD_IN);
  gemm_in_wmma<<<gemmBlk, TPB, 0, stream>>>(poi_ids, cat_ids, feat3, poi_emb, cat_emb,
                                            Wth, xw, N);
  gcn_conv(b_in);
  leaky_copy<<<ewBlk, TPB, 0, stream>>>(x, tb, tot);

  // ---- 5 GcnUnits: (GCN + GraphNorm + leaky + residual), (GAT + same) ----
  for (int l = 0; l < 5; ++l) {
    // GCN half
    convert_wt<<<(CCH * CCH + TPB - 1) / TPB, TPB, 0, stream>>>(gcn_W + (size_t)l * CCH * CCH, Wth, CCH, CCH);
    gemm128_wmma<<<gemmBlk, TPB, 0, stream>>>(x, Wth, xw, N);
    gcn_conv(gcn_b + l * CCH);
    graph_norm_resid(gn_gamma + l * CCH, gn_beta + l * CCH, gn_alpha + l * CCH);
    // GAT half
    convert_wt<<<(CCH * CCH + TPB - 1) / TPB, TPB, 0, stream>>>(gat_W + (size_t)l * CCH * CCH, Wth, CCH, CCH);
    gemm128_wmma<<<gemmBlk, TPB, 0, stream>>>(x, Wth, xw, N);
    gat_alar<<<nodeWBlk, TPB, 0, stream>>>(xw, gat_asrc + l * CCH, gat_adst + l * CCH, al, ar, N);
    gat_premax<<<nodeBlk, TPB, 0, stream>>>(al, ar, emk, N);
    gat_edge_max<<<edgeBlk, TPB, 0, stream>>>(al, ar, src, dst, emk, E);
    gat_den_init<<<nodeBlk, TPB, 0, stream>>>(al, ar, emk, den, N);
    gat_edge_den<<<edgeBlk, TPB, 0, stream>>>(al, ar, src, dst, emk, den, E);
    gat_out_init<<<ewBlk, TPB, 0, stream>>>(tb, xw, gat_b + l * CCH, al, ar, emk, den, N);
    gat_edge_scatter<<<edgeWBlk, TPB, 0, stream>>>(xw, src, dst, al, ar, emk, den, tb, E);
    graph_norm_resid(gn_gamma + l * CCH, gn_beta + l * CCH, gn_alpha + l * CCH);
  }

  // ---- output GCN conv (C -> 1) ----
  wout_matvec<<<nodeWBlk, TPB, 0, stream>>>(x, Wout, al, N);
  sc_init<<<nodeBlk, TPB, 0, stream>>>(svec, al, b_out, dinv, N);
  sc_edge<<<edgeBlk, TPB, 0, stream>>>(al, src, dst, ew, dinv, svec, E);

  // ---- FC head: h = relu(leaky(svec) @ W1 + b1); out = relu(h @ W2 + b2) ----
  h_init<<<1, CCH, 0, stream>>>(hacc, fc_b1);
  fc1_kernel<<<1024, TPB, 0, stream>>>(svec, fc_W1, hacc, N);
  fc2_kernel<<<(PLEN + TPB - 1) / TPB, TPB, 0, stream>>>(hacc, fc_W2, fc_b2, out, PLEN);
}
